// VSONDecoder_32401233281074
// MI455X (gfx1250) — matrix operand order repaired and checked by reading, not compiled, not run on hardware
//
#include <hip/hip_runtime.h>
#include <math.h>

typedef __attribute__((ext_vector_type(16))) _Float16 v16h;
typedef __attribute__((ext_vector_type(8)))  float    v8f;

#define Bn 4
#define Pn 32768
#define Cn 8
#define Fn 32
#define Dn 64
#define Hn 64

#define F32_INF __builtin_inff()

// ---------------- workspace layout (bytes) ----------------
// ZcB  : f16 [B*C][4 ntile][32 lane][16 hw]          @ 0       (131072 B)
// WaB  : f16 [5][2 ktile][4 ntile][32][16]           @ 131072  (40960 B)
// WbB  : f16 [5][2][4][32][16]                       @ 172032  (40960 B)
// bias0: f32 [B*C][H]   (z_inv@W0[65:129] + b0)      @ 212992  (8192 B)
// zinv : f32 [B*C][D]                                @ 221184  (8192 B)

__device__ __forceinline__ v8f wmma_f16(v16h a, v16h b, v8f c) {
  return __builtin_amdgcn_wmma_f32_16x16x32_f16(false, a, false, b, (short)0, c,
                                                false, false);
}

// single v_med3_f32 (no canonicalize pair like fmaxf/ternary lowering)
__device__ __forceinline__ float relu(float x) {
  return __builtin_amdgcn_fmed3f(x, 0.f, F32_INF);
}
__device__ __forceinline__ float fmax2(float a, float b) {
  return __builtin_amdgcn_fmed3f(a, b, F32_INF);
}

// act staging buffer is wave-private; HW executes a wave's LDS ops in order,
// so only a compiler-level ordering barrier is needed (no s_barrier).
__device__ __forceinline__ void wave_sync() {
  asm volatile("" ::: "memory");
  __builtin_amdgcn_wave_barrier();
}

// A-matrix (16xK f16) lane/halfword -> K mapping per ISA 7.12.2:
// lanes 0-15 : K = i (i<8) / i+8 (i>=8);  lanes 16-31: +8
__device__ __forceinline__ void load_A_tiles(const _Float16* act, int lane,
                                             v16h& A0, v16h& A1) {
  const int lm  = lane & 15;
  const int khi = (lane >= 16) ? 8 : 0;
#pragma unroll
  for (int i = 0; i < 16; ++i) {
    int k = ((i < 8) ? i : i + 8) + khi;
    A0[i] = act[lm * Hn + k];
    A1[i] = act[lm * Hn + 32 + k];
  }
}

// one residual block: acc = relu(relu(acc)@Wa+ba)@Wb + bb + acc
__device__ __forceinline__ void resblock(v8f* acc, const _Float16* WaBlk,
                                         const _Float16* WbBlk,
                                         const float* ba_row, const float* bb_row,
                                         _Float16* act, int lane) {
  const int lm    = lane & 15;
  const int rbase = (lane >= 16) ? 8 : 0;
  v8f xs[4];
#pragma unroll
  for (int t = 0; t < 4; ++t) xs[t] = acc[t];

  // stage relu(acc) to LDS (C-layout: vgpr j -> row j+rbase, col t*16+lm)
#pragma unroll
  for (int t = 0; t < 4; ++t)
#pragma unroll
    for (int j = 0; j < 8; ++j)
      act[(j + rbase) * Hn + t * 16 + lm] = (_Float16)relu(acc[t][j]);
  wave_sync();

  v16h A0, A1;
  load_A_tiles(act, lane, A0, A1);
  wave_sync();

  v8f h[4];
#pragma unroll
  for (int t = 0; t < 4; ++t) {
    float bv = ba_row[t * 16 + lm];
    v8f cc;
#pragma unroll
    for (int j = 0; j < 8; ++j) cc[j] = bv;
    v16h B0 = *(const v16h*)(WaBlk + (0 * 4 + t) * 512 + lane * 16);
    v16h B1 = *(const v16h*)(WaBlk + (1 * 4 + t) * 512 + lane * 16);
    cc = wmma_f16(A0, B0, cc);
    cc = wmma_f16(A1, B1, cc);
    h[t] = cc;
  }

#pragma unroll
  for (int t = 0; t < 4; ++t)
#pragma unroll
    for (int j = 0; j < 8; ++j)
      act[(j + rbase) * Hn + t * 16 + lm] = (_Float16)relu(h[t][j]);
  wave_sync();

  load_A_tiles(act, lane, A0, A1);
  wave_sync();

#pragma unroll
  for (int t = 0; t < 4; ++t) {
    float bv = bb_row[t * 16 + lm];
    v8f cc;
#pragma unroll
    for (int j = 0; j < 8; ++j) cc[j] = bv + xs[t][j];  // residual folded into C
    v16h B0 = *(const v16h*)(WbBlk + (0 * 4 + t) * 512 + lane * 16);
    v16h B1 = *(const v16h*)(WbBlk + (1 * 4 + t) * 512 + lane * 16);
    cc = wmma_f16(A0, B0, cc);
    cc = wmma_f16(A1, B1, cc);
    acc[t] = cc;
  }
}

// ---------------- prep kernels ----------------
__global__ void k_zinv(const float* __restrict__ z, const float* __restrict__ Wdir,
                       float* __restrict__ zinv) {
  int idx = blockIdx.x * blockDim.x + threadIdx.x;
  if (idx >= Bn * Cn * Dn) return;
  int d = idx % Dn, bc = idx / Dn;
  const float* zb = z + (size_t)bc * Fn * Dn;
  float acc = 0.f;
  for (int f = 0; f < Fn; ++f) {
    float dot = 0.f;
    for (int e = 0; e < Dn; ++e) dot += zb[f * Dn + e] * Wdir[e * Dn + d];
    acc += zb[f * Dn + d] * dot;
  }
  zinv[idx] = acc;
}

__global__ void k_bias0(const float* __restrict__ zinv, const float* __restrict__ W0,
                        const float* __restrict__ b0, float* __restrict__ bias0) {
  int idx = blockIdx.x * blockDim.x + threadIdx.x;
  if (idx >= Bn * Cn * Hn) return;
  int h = idx % Hn, bc = idx / Hn;
  float acc = b0[h];
  for (int d = 0; d < Dn; ++d)
    acc += zinv[bc * Dn + d] * W0[(1 + Dn + d) * Hn + h];
  bias0[idx] = acc;
}

// Zc[b,c] = z[b,c] (FxD) @ W0[1:1+D] (DxH), swizzled into WMMA B-matrix lane order
__global__ void k_zcb(const float* __restrict__ z, const float* __restrict__ W0,
                      _Float16* __restrict__ ZcB) {
  int idx = blockIdx.x * blockDim.x + threadIdx.x;
  if (idx >= Bn * Cn * 4 * 32 * 16) return;
  int i = idx & 15, lanev = (idx >> 4) & 31, t = (idx >> 9) & 3, bc = idx >> 11;
  int n = t * 16 + (lanev & 15);
  int k = ((i < 8) ? i : i + 8) + ((lanev < 16) ? 0 : 8);  // B-matrix 32x16 K mapping
  const float* zb = z + (size_t)bc * Fn * Dn;
  float acc = 0.f;
  for (int d = 0; d < Dn; ++d) acc += zb[k * Dn + d] * W0[(1 + d) * Hn + n];
  ZcB[idx] = (_Float16)acc;
}

// Wa/Wb (5,H,H) f32 -> f16 B-matrix lane order: [blk][kt][t][lane][i]
__global__ void k_wprep(const float* __restrict__ Wa, const float* __restrict__ Wb,
                        _Float16* __restrict__ WaB, _Float16* __restrict__ WbB) {
  int idx = blockIdx.x * blockDim.x + threadIdx.x;
  if (idx >= 2 * 5 * 2 * 4 * 32 * 16) return;
  int which = idx / 20480;
  int id2   = idx % 20480;
  int i = id2 & 15, lanev = (id2 >> 4) & 31, t = (id2 >> 9) & 3;
  int kt = (id2 >> 11) & 1, blk = id2 >> 12;
  int n = t * 16 + (lanev & 15);
  int k = kt * 32 + ((i < 8) ? i : i + 8) + ((lanev < 16) ? 0 : 8);
  const float* src = which ? Wb : Wa;
  _Float16*    dst = which ? WbB : WaB;
  dst[(((blk * 2 + kt) * 4 + t) * 32 + lanev) * 16 + i] =
      (_Float16)src[(size_t)blk * Hn * Hn + k * Hn + n];
}

// ---------------- main kernel: 8 waves/block, 16 points/wave ----------------
__global__ void __launch_bounds__(256)
k_main(const float* __restrict__ pos, const float* __restrict__ p,
       const float* __restrict__ Wfeat, const float* __restrict__ W0,
       const _Float16* __restrict__ ZcB, const _Float16* __restrict__ WaB,
       const _Float16* __restrict__ WbB, const float* __restrict__ bias0,
       const float* __restrict__ ba, const float* __restrict__ bb,
       const float* __restrict__ Wout, const float* __restrict__ bout,
       float* __restrict__ out) {
  __shared__ _Float16 act_s[8][16 * Hn];

  const int lane  = threadIdx.x & 31;
  const int w     = threadIdx.x >> 5;
  const int tile  = blockIdx.x * 8 + w;
  const int base  = tile * 16;
  const int b     = base / Pn;
  const int lm    = lane & 15;
  const int rbase = (lane >= 16) ? 8 : 0;
  _Float16* act   = act_s[w];

  // warm WGP$/L2 for the prepped weight arrays (global_prefetch_b8)
  __builtin_prefetch(WaB + lane * 64, 0, 3);
  __builtin_prefetch(WbB + lane * 64, 0, 3);
  __builtin_prefetch(ZcB + (size_t)b * Cn * 2048 + lane * 64, 0, 3);

  // --- center + clamp query point (lanes l and l+16 both handle point lm) ---
  int   gp = base + lm;
  float px = p[gp * 3 + 0] - pos[b * 3 + 0];
  float py = p[gp * 3 + 1] - pos[b * 3 + 1];
  float pz = p[gp * 3 + 2] - pos[b * 3 + 2];
  float nr = sqrtf(px * px + py * py + pz * pz);
  if (nr > 0.5f) { float sc = 0.5f / nr; px *= sc; py *= sc; pz *= sc; }

  // --- p_ext directly in A-matrix register layout (16x32, K=F) ---
  v16h  Ape;
  float sp = 0.f;
#pragma unroll
  for (int i = 0; i < 16; ++i) {
    int   k = ((i < 8) ? i : i + 8) + rbase;
    float v = px * Wfeat[k] + py * Wfeat[Fn + k] + pz * Wfeat[2 * Fn + k];
    sp += v * v;
    Ape[i] = (_Float16)v;
  }
  sp += __shfl_xor(sp, 16, 32);  // |p_ext|^2 per point (lane lm holds point lm)

  // per-lane constants for the C-init of the first WMMA
  float w0n[4], srow[8];
#pragma unroll
  for (int t = 0; t < 4; ++t) w0n[t] = W0[t * 16 + lm];  // W0 row 0 (scalar feat)
#pragma unroll
  for (int j = 0; j < 8; ++j) srow[j] = __shfl(sp, j + rbase, 32);

  v8f pooled[4];
#pragma unroll
  for (int t = 0; t < 4; ++t)
#pragma unroll
    for (int j = 0; j < 8; ++j) pooled[t][j] = -3.402823466e38f;

  // --- per-channel: net0 = p_ext@Zc + s*w0row0 + bias0, then block 0, pool ---
  // unroll 1: keep the live set under 256 VGPRs (no s_set_vgpr_msb churn);
  // the scheduler still pipelines B-tile loads across iterations.
#pragma unroll 1
  for (int c = 0; c < Cn; ++c) {
    v8f acc[4];
    const _Float16* zc  = ZcB + (size_t)(b * Cn + c) * 4 * 512;
    const float*    b0c = bias0 + (b * Cn + c) * Hn;
#pragma unroll
    for (int t = 0; t < 4; ++t) {
      float bv = b0c[t * 16 + lm];
      v8f   cc;
#pragma unroll
      for (int j = 0; j < 8; ++j) cc[j] = srow[j] * w0n[t] + bv;
      v16h Bt = *(const v16h*)(zc + t * 512 + lane * 16);
      acc[t]  = wmma_f16(Ape, Bt, cc);
    }
    resblock(acc, WaB, WbB, ba, bb, act, lane);
#pragma unroll
    for (int t = 0; t < 4; ++t)
#pragma unroll
      for (int j = 0; j < 8; ++j) pooled[t][j] = fmax2(pooled[t][j], acc[t][j]);
  }

  // --- blocks 1..4 on pooled activations (serial dependency: keep compact) ---
  v8f acc[4];
#pragma unroll
  for (int t = 0; t < 4; ++t) acc[t] = pooled[t];
#pragma unroll 1
  for (int blk = 1; blk < 5; ++blk)
    resblock(acc, WaB + blk * 4096, WbB + blk * 4096, ba + blk * Hn,
             bb + blk * Hn, act, lane);

  // --- out = relu(net) @ Wout + bout : shuffle-reduce over N ---
  float part[8];
#pragma unroll
  for (int j = 0; j < 8; ++j) part[j] = 0.f;
#pragma unroll
  for (int t = 0; t < 4; ++t) {
    float wv = Wout[t * 16 + lm];
#pragma unroll
    for (int j = 0; j < 8; ++j) part[j] += relu(acc[t][j]) * wv;
  }
#pragma unroll
  for (int m = 1; m < 16; m <<= 1)
#pragma unroll
    for (int j = 0; j < 8; ++j) part[j] += __shfl_xor(part[j], m, 32);

  if (lm == 0) {
    float bo = bout[0];
#pragma unroll
    for (int j = 0; j < 8; ++j) out[base + rbase + j] = part[j] + bo;
  }
}

// ---------------- host launcher ----------------
extern "C" void kernel_launch(void* const* d_in, const int* in_sizes, int n_in,
                              void* d_out, int out_size, void* d_ws, size_t ws_size,
                              hipStream_t stream) {
  const float* z     = (const float*)d_in[0];
  const float* pos   = (const float*)d_in[1];
  const float* p     = (const float*)d_in[2];
  const float* Wfeat = (const float*)d_in[3];
  const float* Wdir  = (const float*)d_in[4];
  const float* W0    = (const float*)d_in[5];
  const float* b0    = (const float*)d_in[6];
  const float* Wa    = (const float*)d_in[7];
  const float* ba    = (const float*)d_in[8];
  const float* Wb    = (const float*)d_in[9];
  const float* bb    = (const float*)d_in[10];
  const float* Wout  = (const float*)d_in[11];
  const float* bout  = (const float*)d_in[12];
  float*       out   = (float*)d_out;

  char*      ws    = (char*)d_ws;
  _Float16*  ZcB   = (_Float16*)(ws + 0);
  _Float16*  WaB   = (_Float16*)(ws + 131072);
  _Float16*  WbB   = (_Float16*)(ws + 172032);
  float*     bias0 = (float*)(ws + 212992);
  float*     zinv  = (float*)(ws + 221184);

  k_zinv<<<8, 256, 0, stream>>>(z, Wdir, zinv);
  k_bias0<<<8, 256, 0, stream>>>(zinv, W0, b0, bias0);
  k_zcb<<<256, 256, 0, stream>>>(z, W0, ZcB);
  k_wprep<<<160, 256, 0, stream>>>(Wa, Wb, WaB, WbB);
  k_main<<<(Bn * Pn) / 128, 256, 0, stream>>>(pos, p, Wfeat, W0, ZcB, WaB, WbB,
                                              bias0, ba, bb, Wout, bout, out);
}
